// E_GCL_33200097198208
// MI455X (gfx1250) — matrix operand order repaired and checked by reading, not compiled, not run on hardware
//
#include <hip/hip_runtime.h>

#define NN   30000
#define NE   480000
#define HIDD 128

typedef __attribute__((ext_vector_type(16))) _Float16     v16h;
typedef __attribute__((ext_vector_type(8)))  float        v8f;
typedef __attribute__((ext_vector_type(4)))  unsigned int v4u;

__device__ __forceinline__ v8f vzero8() { v8f z = {}; return z; }

// A-matrix 16-bit 16x32 fragment (row M = lane&15):
// elems 0..7  <-> K = hi*8 + 0..7
// elems 8..15 <-> K = 16 + hi*8 + 0..7
__device__ __forceinline__ v16h load_fragA(const _Float16* p, int hi) {
  union { v4u q[2]; v16h h; } u;
  u.q[0] = *(const v4u*)(p + hi * 8);
  u.q[1] = *(const v4u*)(p + 16 + hi * 8);
  return u.h;
}
// B-matrix 16-bit 32x16 fragment (col N = lane&15), from transposed [N][K] weights:
// elems 0..7 <-> K = hi*8 + 0..7, elems 8..15 <-> K = 16 + hi*8 + 0..7
__device__ __forceinline__ v16h load_fragB(const _Float16* p, int hi) {
  union { v4u q[2]; v16h h; } u;
  const _Float16* s = p + hi * 8;
  u.q[0] = *(const v4u*)(s);
  u.q[1] = *(const v4u*)(s + 16);
  return u.h;
}
__device__ __forceinline__ v8f wmma16(v16h a, v16h b, v8f c) {
  return __builtin_amdgcn_wmma_f32_16x16x32_f16(false, a, false, b, (short)0, c,
                                                false, false);
}
__device__ __forceinline__ float silu_(float x) {
  return x / (1.0f + __expf(-x));
}

// ---------------------------------------------------------------- prep kernels

// edge_mlp_W [388][128]: rows 0..255 -> W1t f16 [128 n][256 k] (transposed),
// rows 256..259 -> Wc f32 [4][128], rows 260..387 folded w/ prompt into bias1.
__global__ void prep_edge_w(const float* __restrict__ W, const float* __restrict__ b,
                            const float* __restrict__ prompt,
                            _Float16* __restrict__ W1t, float* __restrict__ Wc,
                            float* __restrict__ bias1) {
  int tid = blockIdx.x * blockDim.x + threadIdx.x;
  if (tid < 256 * 128) {
    int n = tid >> 8, k = tid & 255;
    W1t[n * 256 + k] = (_Float16)W[k * 128 + n];
  }
  int t2 = tid - 256 * 128;
  if (t2 >= 0 && t2 < 512) {
    int t = t2 >> 7, n = t2 & 127;
    Wc[t * 128 + n] = W[(256 + t) * 128 + n];
  }
  int t3 = tid - (256 * 128 + 512);
  if (t3 >= 0 && t3 < 128) {
    float s = b[t3];
    for (int q = 0; q < 128; ++q) s += prompt[q] * W[(260 + q) * 128 + t3];
    bias1[t3] = s;
  }
}

// 128x128 f32 -> f16 transposed
__global__ void prep_tr128(const float* __restrict__ W, _Float16* __restrict__ Wt) {
  int tid = blockIdx.x * blockDim.x + threadIdx.x;
  if (tid < 128 * 128) {
    int n = tid >> 7, k = tid & 127;
    Wt[n * 128 + k] = (_Float16)W[k * 128 + n];
  }
}

// Hm[r][n] = sum_k rule[k, r/A, n/32] * W[k, r%A, n%32]
__device__ __forceinline__ float hm_val(const float* rule, const float* W, int A,
                                        int r, int n) {
  int i = r / A, a = r % A, j = n >> 5, bc = n & 31;
  float s = 0.f;
#pragma unroll
  for (int k = 0; k < 4; ++k)
    s += rule[k * 16 + i * 4 + j] * W[(k * A + a) * 32 + bc];
  return s;
}

// Build f16 transposed Hm_t[n][rsplit] for WMMA rows < rsplit; fold rows
// [rsplit, rtotal) (the broadcast prompt block) with prompt into bias_out.
__global__ void prep_phm(const float* __restrict__ rule, const float* __restrict__ W,
                         const float* __restrict__ bin, const float* __restrict__ prompt,
                         int A, int rsplit, int rtotal,
                         _Float16* __restrict__ Hm_t, float* __restrict__ bias_out) {
  int tid = blockIdx.x * blockDim.x + threadIdx.x;
  if (tid < 128 * rsplit) {
    int n = tid / rsplit, r = tid % rsplit;
    Hm_t[n * rsplit + r] = (_Float16)hm_val(rule, W, A, r, n);
  } else if (tid < 128 * rsplit + 128) {
    int n = tid - 128 * rsplit;
    float s = bin[n];
    for (int r = rsplit; r < rtotal; ++r)
      s += prompt[r - rsplit] * hm_val(rule, W, A, r, n);
    bias_out[n] = s;
  }
}

// h f32 -> h16 f16; zero agg [N][128] and agg_c [N][3] (fresh every launch)
__global__ void prep_h_zero(const float* __restrict__ h, _Float16* __restrict__ h16,
                            float* __restrict__ agg, float* __restrict__ agg_c) {
  int tid = blockIdx.x * blockDim.x + threadIdx.x;
  if (tid < NN * HIDD) {
    h16[tid] = (_Float16)h[tid];
    agg[tid] = 0.0f;
  }
  if (tid < NN * 3) agg_c[tid] = 0.0f;
}

__global__ void conv_agg(const float* __restrict__ agg, _Float16* __restrict__ agg16) {
  int tid = blockIdx.x * blockDim.x + threadIdx.x;
  if (tid < NN * HIDD) agg16[tid] = (_Float16)agg[tid];
}

// ---------------------------------------------------------------- edge kernel
// Per wave: 16 edges. Fused chain:
//   m  = silu(h16[row]‖h16[col] @ W1t  + [radial,ea]@Wc + bias1)   (K=256 WMMA + rank-4)
//   ef = silu(m @ Hm_e_t + b2)            -> atomicAdd into agg    (K=128 WMMA)
//   u  = silu(ef @ Hm_ce_t + bce)
//   acc_e = u . cew  (LDS ds_add_f32 reduce), trans scatter to agg_c
__global__ void __launch_bounds__(256) edge_kernel(
    const _Float16* __restrict__ h16, const long long* __restrict__ eidx,
    const float* __restrict__ coord, const float* __restrict__ eattr,
    const _Float16* __restrict__ W1t, const float* __restrict__ Wc,
    const float* __restrict__ bias1, const _Float16* __restrict__ HmE,
    const float* __restrict__ b2, const _Float16* __restrict__ HmCE,
    const float* __restrict__ bce, const float* __restrict__ cew,
    float* __restrict__ agg, float* __restrict__ agg_c) {
  __shared__ __attribute__((aligned(16))) _Float16 mbuf[8][16][HIDD];
  __shared__ float rad_s[8][16];
  __shared__ float ea_s[8][16][3];
  __shared__ float cd_s[8][16][3];
  __shared__ int row_s[8][16];
  __shared__ int col_s[8][16];
  __shared__ float accE[8][16];

  const int wave = threadIdx.x >> 5, lane = threadIdx.x & 31;
  const int hi = lane >> 4, li = lane & 15;
  const int tile = blockIdx.x * 8 + wave;
  const int e0 = tile * 16;

  if (lane < 16) {
    int e = e0 + lane;
    int r = (int)eidx[e];
    int c = (int)eidx[NE + e];
    row_s[wave][lane] = r;
    col_s[wave][lane] = c;
    float dx = coord[3 * r + 0] - coord[3 * c + 0];
    float dy = coord[3 * r + 1] - coord[3 * c + 1];
    float dz = coord[3 * r + 2] - coord[3 * c + 2];
    float rad = dx * dx + dy * dy + dz * dz;
    rad_s[wave][lane] = rad;
    float inv = 1.0f / fmaxf(sqrtf(rad), 1e-12f);
    cd_s[wave][lane][0] = dx * inv;
    cd_s[wave][lane][1] = dy * inv;
    cd_s[wave][lane][2] = dz * inv;
    ea_s[wave][lane][0] = eattr[3 * e + 0];
    ea_s[wave][lane][1] = eattr[3 * e + 1];
    ea_s[wave][lane][2] = eattr[3 * e + 2];
    accE[wave][lane] = 0.0f;
  }
  // same-wave LDS is in-order; no barrier needed (all LDS is wave-private)

  const int r0 = row_s[wave][li];
  const int c0 = col_s[wave][li];
  const _Float16* Ar = h16 + (long)r0 * HIDD;
  const _Float16* Ac = h16 + (long)c0 * HIDD;

  // --- GEMM1: K=256 (h[row]‖h[col]) ---
  v8f acc[8];
#pragma unroll
  for (int i = 0; i < 8; ++i) acc[i] = vzero8();
#pragma unroll
  for (int kc = 0; kc < 8; ++kc) {
    const _Float16* src = (kc < 4) ? (Ar + kc * 32) : (Ac + (kc - 4) * 32);
    v16h a = load_fragA(src, hi);
#pragma unroll
    for (int nt = 0; nt < 8; ++nt) {
      v16h b = load_fragB(W1t + (nt * 16 + li) * 256 + kc * 32, hi);
      acc[nt] = wmma16(a, b, acc[nt]);
    }
  }

  // per-row extras for this lane's 8 C rows (M = hi*8 + rr)
  float rad8[8], ex[8], ey[8], ez[8];
  int rw8[8];
#pragma unroll
  for (int rr = 0; rr < 8; ++rr) {
    int M = hi * 8 + rr;
    rad8[rr] = rad_s[wave][M];
    ex[rr] = ea_s[wave][M][0];
    ey[rr] = ea_s[wave][M][1];
    ez[rr] = ea_s[wave][M][2];
    rw8[rr] = row_s[wave][M];
  }
#pragma unroll
  for (int nt = 0; nt < 8; ++nt) {
    int n = nt * 16 + li;
    float bb = bias1[n];
    float w0 = Wc[n], w1 = Wc[128 + n], w2 = Wc[256 + n], w3 = Wc[384 + n];
#pragma unroll
    for (int rr = 0; rr < 8; ++rr) {
      float x = acc[nt][rr] + bb + rad8[rr] * w0 + ex[rr] * w1 + ey[rr] * w2 +
                ez[rr] * w3;
      mbuf[wave][hi * 8 + rr][n] = (_Float16)silu_(x);
    }
  }

  const _Float16* Am = &mbuf[wave][0][0];

  // --- GEMM2: PHM edge, K=128 ---
#pragma unroll
  for (int i = 0; i < 8; ++i) acc[i] = vzero8();
#pragma unroll
  for (int kc = 0; kc < 4; ++kc) {
    v16h a = load_fragA(Am + li * HIDD + kc * 32, hi);
#pragma unroll
    for (int nt = 0; nt < 8; ++nt) {
      v16h b = load_fragB(HmE + (nt * 16 + li) * 128 + kc * 32, hi);
      acc[nt] = wmma16(a, b, acc[nt]);
    }
  }
#pragma unroll
  for (int nt = 0; nt < 8; ++nt) {
    int n = nt * 16 + li;
    float bb = b2[n];
#pragma unroll
    for (int rr = 0; rr < 8; ++rr) {
      float x = silu_(acc[nt][rr] + bb);          // edge_feat
      atomicAdd(&agg[rw8[rr] * HIDD + n], x);     // segment_sum over row
      mbuf[wave][hi * 8 + rr][n] = (_Float16)x;   // stage for ce GEMM
    }
  }

  // --- GEMM3: PHM ce, K=128, then dot with coord_edge_W ---
#pragma unroll
  for (int i = 0; i < 8; ++i) acc[i] = vzero8();
#pragma unroll
  for (int kc = 0; kc < 4; ++kc) {
    v16h a = load_fragA(Am + li * HIDD + kc * 32, hi);
#pragma unroll
    for (int nt = 0; nt < 8; ++nt) {
      v16h b = load_fragB(HmCE + (nt * 16 + li) * 128 + kc * 32, hi);
      acc[nt] = wmma16(a, b, acc[nt]);
    }
  }
#pragma unroll
  for (int nt = 0; nt < 8; ++nt) {
    int n = nt * 16 + li;
    float bb = bce[n], w = cew[n];
#pragma unroll
    for (int rr = 0; rr < 8; ++rr) {
      float u = silu_(acc[nt][rr] + bb);
      atomicAdd(&accE[wave][hi * 8 + rr], u * w);  // ds_add_f32 reduce over N
    }
  }
  // same-wave LDS atomics complete in-order before the load below
  if (lane < 16) {
    float ae = accE[wave][lane];
    int rr2 = row_s[wave][lane];
    atomicAdd(&agg_c[rr2 * 3 + 0], cd_s[wave][lane][0] * ae);
    atomicAdd(&agg_c[rr2 * 3 + 1], cd_s[wave][lane][1] * ae);
    atomicAdd(&agg_c[rr2 * 3 + 2], cd_s[wave][lane][2] * ae);
  }
}

// ---------------------------------------------------------------- node kernel
// Per wave: 16 nodes. t = [h‖agg]@Hm_n1t + bias_n; h_out = h + silu(t)@Wn2t + nb
__global__ void __launch_bounds__(256) node_kernel(
    const _Float16* __restrict__ h16, const _Float16* __restrict__ agg16,
    const _Float16* __restrict__ HmN, const float* __restrict__ bias_n,
    const _Float16* __restrict__ Wn2t, const float* __restrict__ nb,
    const float* __restrict__ h, float* __restrict__ hout,
    _Float16* __restrict__ hout16) {
  __shared__ __attribute__((aligned(16))) _Float16 sbuf[8][16][HIDD];
  const int wave = threadIdx.x >> 5, lane = threadIdx.x & 31;
  const int tile = blockIdx.x * 8 + wave;
  if (tile >= NN / 16) return;  // wave-uniform, no barriers in this kernel
  const int n0 = tile * 16;
  const int hi = lane >> 4, li = lane & 15;

  const _Float16* Ah = h16 + (long)(n0 + li) * HIDD;
  const _Float16* Ag = agg16 + (long)(n0 + li) * HIDD;

  v8f acc[8];
#pragma unroll
  for (int i = 0; i < 8; ++i) acc[i] = vzero8();
#pragma unroll
  for (int kc = 0; kc < 8; ++kc) {
    const _Float16* src = (kc < 4) ? (Ah + kc * 32) : (Ag + (kc - 4) * 32);
    v16h a = load_fragA(src, hi);
#pragma unroll
    for (int nt = 0; nt < 8; ++nt) {
      v16h b = load_fragB(HmN + (nt * 16 + li) * 256 + kc * 32, hi);
      acc[nt] = wmma16(a, b, acc[nt]);
    }
  }
#pragma unroll
  for (int nt = 0; nt < 8; ++nt) {
    int n = nt * 16 + li;
    float bb = bias_n[n];
#pragma unroll
    for (int rr = 0; rr < 8; ++rr)
      sbuf[wave][hi * 8 + rr][n] = (_Float16)silu_(acc[nt][rr] + bb);
  }

  const _Float16* As = &sbuf[wave][0][0];
#pragma unroll
  for (int i = 0; i < 8; ++i) acc[i] = vzero8();
#pragma unroll
  for (int kc = 0; kc < 4; ++kc) {
    v16h a = load_fragA(As + li * HIDD + kc * 32, hi);
#pragma unroll
    for (int nt = 0; nt < 8; ++nt) {
      v16h b = load_fragB(Wn2t + (nt * 16 + li) * 128 + kc * 32, hi);
      acc[nt] = wmma16(a, b, acc[nt]);
    }
  }
#pragma unroll
  for (int nt = 0; nt < 8; ++nt) {
    int n = nt * 16 + li;
    float bb = nb[n];
#pragma unroll
    for (int rr = 0; rr < 8; ++rr) {
      int node = n0 + hi * 8 + rr;
      float ho = h[node * HIDD + n] + acc[nt][rr] + bb;  // residual
      hout[node * HIDD + n] = ho;
      hout16[node * HIDD + n] = (_Float16)ho;
    }
  }
}

// ---------------------------------------------------------------- cp kernel
// acc_p = silu(h_out @ Hm_cp + bias_cp) . cpw ; acc = agg_c * acc_p
__global__ void __launch_bounds__(256) cp_kernel(
    const _Float16* __restrict__ hout16, const _Float16* __restrict__ HmCP,
    const float* __restrict__ bias_cp, const float* __restrict__ cpw,
    const float* __restrict__ agg_c, float* __restrict__ acc_out) {
  __shared__ float accP[8][16];
  const int wave = threadIdx.x >> 5, lane = threadIdx.x & 31;
  const int tile = blockIdx.x * 8 + wave;
  if (tile >= NN / 16) return;
  const int n0 = tile * 16;
  const int hi = lane >> 4, li = lane & 15;

  if (lane < 16) accP[wave][lane] = 0.0f;

  const _Float16* Ah = hout16 + (long)(n0 + li) * HIDD;
  v8f acc[8];
#pragma unroll
  for (int i = 0; i < 8; ++i) acc[i] = vzero8();
#pragma unroll
  for (int kc = 0; kc < 4; ++kc) {
    v16h a = load_fragA(Ah + kc * 32, hi);
#pragma unroll
    for (int nt = 0; nt < 8; ++nt) {
      v16h b = load_fragB(HmCP + (nt * 16 + li) * 128 + kc * 32, hi);
      acc[nt] = wmma16(a, b, acc[nt]);
    }
  }
#pragma unroll
  for (int nt = 0; nt < 8; ++nt) {
    int n = nt * 16 + li;
    float bb = bias_cp[n], w = cpw[n];
#pragma unroll
    for (int rr = 0; rr < 8; ++rr) {
      float u = silu_(acc[nt][rr] + bb);
      atomicAdd(&accP[wave][hi * 8 + rr], u * w);
    }
  }
  if (lane < 16) {
    int node = n0 + lane;
    float ap = accP[wave][lane];
    acc_out[node * 3 + 0] = agg_c[node * 3 + 0] * ap;
    acc_out[node * 3 + 1] = agg_c[node * 3 + 1] * ap;
    acc_out[node * 3 + 2] = agg_c[node * 3 + 2] * ap;
  }
}

// ---------------------------------------------------------------- launch
extern "C" void kernel_launch(void* const* d_in, const int* in_sizes, int n_in,
                              void* d_out, int out_size, void* d_ws, size_t ws_size,
                              hipStream_t stream) {
  const float* h      = (const float*)d_in[0];
  const long long* ei = (const long long*)d_in[1];
  const float* coord  = (const float*)d_in[2];
  const float* eattr  = (const float*)d_in[3];
  const float* prompt = (const float*)d_in[4];
  const float* W_edge = (const float*)d_in[5];
  const float* b_edge = (const float*)d_in[6];
  const float* r_pe   = (const float*)d_in[7];
  const float* W_pe   = (const float*)d_in[8];
  const float* b_pe   = (const float*)d_in[9];
  const float* r_pn   = (const float*)d_in[10];
  const float* W_pn   = (const float*)d_in[11];
  const float* b_pn   = (const float*)d_in[12];
  const float* W_nm   = (const float*)d_in[13];
  const float* b_nm   = (const float*)d_in[14];
  const float* r_ce   = (const float*)d_in[15];
  const float* W_ce   = (const float*)d_in[16];
  const float* b_ce   = (const float*)d_in[17];
  const float* cew    = (const float*)d_in[18];
  const float* r_cp   = (const float*)d_in[19];
  const float* W_cp   = (const float*)d_in[20];
  const float* b_cp   = (const float*)d_in[21];
  const float* cpw    = (const float*)d_in[22];

  char* ws = (char*)d_ws;
  size_t o = 0;
  auto carve = [&](size_t bytes) {
    char* p = ws + o;
    o += (bytes + 255) & ~(size_t)255;
    return p;
  };
  _Float16* h16    = (_Float16*)carve((size_t)NN * HIDD * 2);
  _Float16* hout16 = (_Float16*)carve((size_t)NN * HIDD * 2);
  float* agg       = (float*)carve((size_t)NN * HIDD * 4);
  _Float16* agg16  = (_Float16*)carve((size_t)NN * HIDD * 2);
  float* agg_c     = (float*)carve((size_t)NN * 3 * 4);
  _Float16* W1t    = (_Float16*)carve(128 * 256 * 2);
  float* Wc        = (float*)carve(4 * 128 * 4);
  float* bias1     = (float*)carve(128 * 4);
  _Float16* HmE    = (_Float16*)carve(128 * 128 * 2);
  float* biasE     = (float*)carve(128 * 4);
  _Float16* HmN    = (_Float16*)carve(128 * 256 * 2);
  float* biasN     = (float*)carve(128 * 4);
  _Float16* HmCE   = (_Float16*)carve(128 * 128 * 2);
  float* biasCE    = (float*)carve(128 * 4);
  _Float16* HmCP   = (_Float16*)carve(128 * 128 * 2);
  float* biasCP    = (float*)carve(128 * 4);
  _Float16* Wn2t   = (_Float16*)carve(128 * 128 * 2);

  float* out       = (float*)d_out;
  float* hout      = out;                       // [N,128]
  float* coord_out = out + (size_t)NN * HIDD;   // [N,3]
  float* acc_out   = coord_out + (size_t)NN * 3;

  // weight / bias prep (cheap; rebuilt every launch — no cross-call state)
  prep_edge_w<<<(256 * 128 + 512 + 128 + 255) / 256, 256, 0, stream>>>(
      W_edge, b_edge, prompt, W1t, Wc, bias1);
  prep_tr128<<<64, 256, 0, stream>>>(W_nm, Wn2t);
  prep_phm<<<(128 * 128 + 128 + 255) / 256, 256, 0, stream>>>(
      r_pe, W_pe, b_pe, prompt, 32, 128, 128, HmE, biasE);
  prep_phm<<<(128 * 256 + 128 + 255) / 256, 256, 0, stream>>>(
      r_pn, W_pn, b_pn, prompt, 96, 256, 384, HmN, biasN);
  prep_phm<<<(128 * 128 + 128 + 255) / 256, 256, 0, stream>>>(
      r_ce, W_ce, b_ce, prompt, 64, 128, 256, HmCE, biasCE);
  prep_phm<<<(128 * 128 + 128 + 255) / 256, 256, 0, stream>>>(
      r_cp, W_cp, b_cp, prompt, 64, 128, 256, HmCP, biasCP);
  prep_h_zero<<<(NN * HIDD + 255) / 256, 256, 0, stream>>>(h, h16, agg, agg_c);

  // fused edge chain: 480000 edges / 16 per wave / 8 waves per block
  edge_kernel<<<NE / (16 * 8), 256, 0, stream>>>(
      h16, ei, coord, eattr, W1t, Wc, bias1, HmE, biasE, HmCE, biasCE, cew,
      agg, agg_c);

  conv_agg<<<(NN * HIDD + 255) / 256, 256, 0, stream>>>(agg, agg16);

  node_kernel<<<(NN / 16 + 7) / 8, 256, 0, stream>>>(
      h16, agg16, HmN, biasN, Wn2t, b_nm, h, hout, hout16);

  cp_kernel<<<(NN / 16 + 7) / 8, 256, 0, stream>>>(
      hout16, HmCP, biasCP, cpw, agg_c, acc_out);

  // coord passes through unchanged
  hipMemcpyAsync(coord_out, coord, (size_t)NN * 3 * 4, hipMemcpyDeviceToDevice,
                 stream);
}
